// HAN_20933670601400
// MI455X (gfx1250) — matrix operand order repaired and checked by reading, not compiled, not run on hardware
//
#include <hip/hip_runtime.h>
#include <hip/hip_bf16.h>

// ---------------------------------------------------------------------------
// Types for CDNA5 WMMA (wave32, v_wmma_f32_16x16x32_bf16)
// ---------------------------------------------------------------------------
typedef __attribute__((ext_vector_type(16))) __bf16          v16bf;
typedef __attribute__((ext_vector_type(8)))  float           v8f;
typedef __attribute__((ext_vector_type(16))) unsigned short  v16u;

__device__ __forceinline__ v16bf as_bf16(v16u v) {
    return __builtin_bit_cast(v16bf, v);
}

__device__ __forceinline__ unsigned short f2bf(float f) {
    unsigned int u = __float_as_uint(f);
    unsigned int r = u + 0x7fffu + ((u >> 16) & 1u);   // round-to-nearest-even
    return (unsigned short)(r >> 16);
}
__device__ __forceinline__ float bf2f(unsigned short u) {
    return __uint_as_float(((unsigned int)u) << 16);
}

__device__ __forceinline__ void atomicMaxF(float* addr, float v) {
    if (v >= 0.0f) atomicMax((int*)addr, __float_as_int(v));
    else           atomicMin((unsigned int*)addr, (unsigned int)__float_as_int(v));
}

// ---------------------------------------------------------------------------
// Generic WMMA GEMM:  C[rows,N] = A[rows,K] (f32 or bf16) * B[K,N] (bf16)
// Block: 256 threads = 8 waves; block tile 16 rows x 128 cols;
// each wave owns one 16x16 C tile; K staged in 64-wide LDS panels.
// B is staged PRE-SWIZZLED into a per-lane fragment order (k = 16h + j, as A) so every
// B fragment is one contiguous 32B/lane read (2x ds_load_b128, no gathers).
// EPI==0: C = A*B (+bias[col]); C routed through LDS -> coalesced b128 stores.
// EPI==1: per-row attention logit: atomicAdd(wlog[row%P], tanh(A*B+b1) . w2)
// ---------------------------------------------------------------------------
template<bool AF32, int EPI>
__global__ __launch_bounds__(256)
void wmma_gemm(const void* __restrict__ Aptr, const unsigned short* __restrict__ B,
               float* __restrict__ C, int K, int N,
               const float* __restrict__ bias, const float* __restrict__ w2,
               float* __restrict__ wlog, int P) {
    __shared__ __align__(32) unsigned short As[16][64];
    // [ksHalf][waveTile][lane][j]: lane = (col&15)|(k&16), j = k&15
    __shared__ __align__(32) unsigned short Bs[2][8][32][16];
    __shared__ __align__(16) float Hs[16][128];

    const int tid  = threadIdx.x;
    const int lane = tid & 31;
    const int wv   = tid >> 5;
    const int rowBase = blockIdx.y * 16;
    const int colBase = blockIdx.x * 128;
    const int tileCol = colBase + wv * 16;
    const bool active = (tileCol < N);          // wave-uniform => EXEC all-ones

    v8f c = {0.f,0.f,0.f,0.f,0.f,0.f,0.f,0.f};

    for (int k0 = 0; k0 < K; k0 += 64) {
        // ---- stage A panel (16 x 64), 4 elems / thread ----
        {
            int idx = tid * 4, r = idx >> 6, cc = idx & 63;
            if (AF32) {
                const float* A = (const float*)Aptr;
                const float* p = A + (size_t)(rowBase + r) * K + (k0 + cc);
                As[r][cc+0] = f2bf(p[0]); As[r][cc+1] = f2bf(p[1]);
                As[r][cc+2] = f2bf(p[2]); As[r][cc+3] = f2bf(p[3]);
            } else {
                const unsigned short* A = (const unsigned short*)Aptr;
                const unsigned short* p = A + (size_t)(rowBase + r) * K + (k0 + cc);
                As[r][cc+0] = p[0]; As[r][cc+1] = p[1];
                As[r][cc+2] = p[2]; As[r][cc+3] = p[3];
            }
        }
        // ---- stage B panel (64 x 128) swizzled to fragment order ----
        for (int it = tid; it < 2048; it += 256) {      // 2048 groups of 4 cols
            int i = it * 4;
            int r = i >> 7, cc = i & 127;
            int gc = colBase + cc;
            unsigned short vals[4];
            if (gc + 3 < N) {
                const unsigned int* p =
                    (const unsigned int*)(B + (size_t)(k0 + r) * N + gc);
                unsigned int w0 = p[0], w1 = p[1];
                vals[0] = (unsigned short)(w0 & 0xffffu);
                vals[1] = (unsigned short)(w0 >> 16);
                vals[2] = (unsigned short)(w1 & 0xffffu);
                vals[3] = (unsigned short)(w1 >> 16);
            } else {
                #pragma unroll
                for (int q = 0; q < 4; ++q)
                    vals[q] = (gc + q < N) ? B[(size_t)(k0 + r) * N + gc + q]
                                           : (unsigned short)0;
            }
            const int ksh = r >> 5, j = r & 15, rk = r & 16, wt = cc >> 4;
            #pragma unroll
            for (int q = 0; q < 4; ++q)
                Bs[ksh][wt][((cc + q) & 15) | rk][j] = vals[q];
        }
        __syncthreads();

        if (active) {
            #pragma unroll
            for (int ks = 0; ks < 64; ks += 32) {
                v16u av;
                const int ar = lane & 15;
                const int ak = (lane & 16) ? 16 : 0;    // contiguous k = 16h + s (matches Bs)
                #pragma unroll
                for (int j = 0; j < 8; ++j) {
                    av[j]     = As[ar][ks + ak + j];
                    av[8 + j] = As[ar][ks + ak + 8 + j];
                }
                v16u bv = *(const v16u*)&Bs[ks >> 5][wv][lane][0];
                c = __builtin_amdgcn_wmma_f32_16x16x32_bf16(
                        false, as_bf16(av), false, as_bf16(bv),
                        (short)0, c, false, false);
            }
        }
        __syncthreads();
    }

    const int n    = lane & 15;
    const int roff = (lane & 16) ? 8 : 0;               // C tile: M = i (+8 hi lanes)
    if (EPI == 0) {
        // stash C tiles in LDS, then store 16x128 block with coalesced b128s
        if (active) {
            #pragma unroll
            for (int i = 0; i < 8; ++i)
                Hs[roff + i][(wv << 4) + n] = c[i];
        }
        __syncthreads();
        const int row = tid >> 4;
        const int cg  = (tid & 15) << 3;                 // 8 cols / thread
        const int gcol = colBase + cg;
        if (gcol < N) {                                  // N % 8 == 0 always here
            float v[8];
            #pragma unroll
            for (int q = 0; q < 8; ++q) v[q] = Hs[row][cg + q];
            if (bias) {
                #pragma unroll
                for (int q = 0; q < 8; ++q) v[q] += bias[gcol + q];
            }
            float* dstp = C + (size_t)(rowBase + row) * N + gcol;
            *(float4*)(dstp)     = make_float4(v[0], v[1], v[2], v[3]);
            *(float4*)(dstp + 4) = make_float4(v[4], v[5], v[6], v[7]);
        }
    } else {
        if (active) {                                    // N==128 here
            int gc = tileCol + n;
            #pragma unroll
            for (int i = 0; i < 8; ++i)
                Hs[roff + i][gc] = tanhf(c[i] + bias[gc]);
        }
        __syncthreads();
        if (tid < 16) {
            float acc = 0.f;
            #pragma unroll 8
            for (int j = 0; j < 128; ++j) acc += Hs[tid][j] * w2[j];
            atomicAdd(&wlog[(rowBase + tid) % P], acc);
        }
    }
}

// ---------------------------------------------------------------------------
// Small helper kernels
// ---------------------------------------------------------------------------
__global__ void fill_u32(unsigned int* p, size_t n, unsigned int v) {
    for (size_t i = blockIdx.x * (size_t)blockDim.x + threadIdx.x; i < n;
         i += (size_t)gridDim.x * blockDim.x) p[i] = v;
}

__global__ void cast_bf16(const float* __restrict__ s, unsigned short* __restrict__ d, size_t n) {
    for (size_t i = blockIdx.x * (size_t)blockDim.x + threadIdx.x; i < n;
         i += (size_t)gridDim.x * blockDim.x) d[i] = f2bf(s[i]);
}

// el/er: [NV,H] dot over O=64 of proj row with al/ar
__global__ void elr_kernel(const float* __restrict__ proj, const float* __restrict__ al,
                           const float* __restrict__ ar, float* __restrict__ el,
                           float* __restrict__ er, int NVH) {
    for (int i = blockIdx.x * blockDim.x + threadIdx.x; i < NVH;
         i += gridDim.x * blockDim.x) {
        int nid = i >> 3, h = i & 7;
        const float* pr = proj + ((size_t)nid << 9) + (h << 6);
        const float* a  = al + (h << 6);
        const float* b  = ar + (h << 6);
        float sl = 0.f, sr = 0.f;
        #pragma unroll 8
        for (int o = 0; o < 64; ++o) { float p = pr[o]; sl += p * a[o]; sr += p * b[o]; }
        el[i] = sl; er[i] = sr;
    }
}

__device__ __forceinline__ float lrelu02(float v) { return v > 0.f ? v : 0.2f * v; }

__global__ void edge_max(const int* __restrict__ s, const int* __restrict__ d,
                         const float* __restrict__ el, const float* __restrict__ er,
                         float* __restrict__ mx, long long EH) {
    for (long long i = blockIdx.x * (long long)blockDim.x + threadIdx.x; i < EH;
         i += (long long)gridDim.x * blockDim.x) {
        int e = (int)(i >> 3), h = (int)(i & 7);
        int ss = s[e], dd = d[e];
        atomicMaxF(&mx[dd * 8 + h], lrelu02(el[ss * 8 + h] + er[dd * 8 + h]));
    }
}

__global__ void edge_den(const int* __restrict__ s, const int* __restrict__ d,
                         const float* __restrict__ el, const float* __restrict__ er,
                         const float* __restrict__ mx, float* __restrict__ den, long long EH) {
    for (long long i = blockIdx.x * (long long)blockDim.x + threadIdx.x; i < EH;
         i += (long long)gridDim.x * blockDim.x) {
        int e = (int)(i >> 3), h = (int)(i & 7);
        int ss = s[e], dd = d[e];
        float v = lrelu02(el[ss * 8 + h] + er[dd * 8 + h]);
        atomicAdd(&den[dd * 8 + h], __expf(v - mx[dd * 8 + h]));
    }
}

// one wave per edge: gout[d,h,:] += alpha * proj[s,h,:]
__global__ __launch_bounds__(256)
void edge_scatter(const int* __restrict__ s, const int* __restrict__ d,
                  const float* __restrict__ el, const float* __restrict__ er,
                  const float* __restrict__ mx, const float* __restrict__ den,
                  const float* __restrict__ proj, float* __restrict__ gout, int E) {
    const int lane = threadIdx.x & 31;
    const int wv   = threadIdx.x >> 5;
    for (int e = blockIdx.x * 8 + wv; e < E; e += gridDim.x * 8) {
        int ss = s[e], dd = d[e];
        const float* ps = proj + ((size_t)ss << 9);
        float*       po = gout + ((size_t)dd << 9);
        #pragma unroll
        for (int h = 0; h < 8; ++h) {
            float v = lrelu02(el[ss * 8 + h] + er[dd * 8 + h]);
            float alpha = __expf(v - mx[dd * 8 + h]) / den[dd * 8 + h];
            #pragma unroll
            for (int o = lane; o < 64; o += 32)
                atomicAdd(&po[(h << 6) + o], alpha * ps[(h << 6) + o]);
        }
    }
}

// z_bf16[n, m, :] = bf16(elu(gout[n,:] + b[:]))
__global__ void gat_finalize(const float* __restrict__ gout, const float* __restrict__ b,
                             unsigned short* __restrict__ zbf, int m, size_t NVE) {
    for (size_t i = blockIdx.x * (size_t)blockDim.x + threadIdx.x; i < NVE;
         i += (size_t)gridDim.x * blockDim.x) {
        size_t nid = i >> 9; int j = (int)(i & 511);
        float v = gout[i] + b[j];
        v = v > 0.f ? v : (__expf(v) - 1.f);            // elu
        zbf[(nid * 3 + m) * 512 + j] = f2bf(v);
    }
}

__global__ void beta_kernel(const float* __restrict__ wlog, float* __restrict__ beta,
                            int P, float invCnt) {
    if (threadIdx.x == 0 && blockIdx.x == 0) {
        float x[4], m = -1e30f;
        for (int p = 0; p < P; ++p) { x[p] = wlog[p] * invCnt; m = fmaxf(m, x[p]); }
        float sum = 0.f;
        for (int p = 0; p < P; ++p) { x[p] = __expf(x[p] - m); sum += x[p]; }
        for (int p = 0; p < P; ++p) beta[p] = x[p] / sum;
    }
}

// ztbf[valid_nodes[n], t, :] = bf16( sum_m beta[m] * zbf[n,m,:] )
__global__ void emb_scatter(const unsigned short* __restrict__ zbf,
                            const float* __restrict__ beta, const int* __restrict__ vn,
                            unsigned short* __restrict__ ztbf, int t, size_t NVE) {
    for (size_t i = blockIdx.x * (size_t)blockDim.x + threadIdx.x; i < NVE;
         i += (size_t)gridDim.x * blockDim.x) {
        size_t nid = i >> 9; int j = (int)(i & 511);
        float e = 0.f;
        #pragma unroll
        for (int m = 0; m < 3; ++m) e += beta[m] * bf2f(zbf[(nid * 3 + m) * 512 + j]);
        int node = vn[nid];
        ztbf[((size_t)node * 4 + t) * 512 + j] = f2bf(e);
    }
}

// tempbf[n,:] = bf16( sum_t betaT[t] * ztbf[n,t,:] )
__global__ void temporal_mix(const unsigned short* __restrict__ ztbf,
                             const float* __restrict__ betaT,
                             unsigned short* __restrict__ tempbf, size_t NNE) {
    for (size_t i = blockIdx.x * (size_t)blockDim.x + threadIdx.x; i < NNE;
         i += (size_t)gridDim.x * blockDim.x) {
        size_t nid = i >> 9; int j = (int)(i & 511);
        float v = 0.f;
        #pragma unroll
        for (int t = 0; t < 4; ++t) v += betaT[t] * bf2f(ztbf[(nid * 4 + t) * 512 + j]);
        tempbf[i] = f2bf(v);
    }
}

// ---------------------------------------------------------------------------
// Host-side orchestration
// ---------------------------------------------------------------------------
extern "C" void kernel_launch(void* const* d_in, const int* in_sizes, int n_in,
                              void* d_out, int out_size, void* d_ws, size_t ws_size,
                              hipStream_t stream) {
    (void)n_in; (void)ws_size;
    const int T_ = 4, M_ = 3, IN_ = 256, H_ = 8, EMB_ = 512, HID_ = 128, OUT_ = 16;
    const int NV_ = in_sizes[3] / T_;
    const int E_  = (int)((long long)in_sizes[1] / (T_ * M_));
    const int NN_ = out_size / OUT_;

    const float* features = (const float*)d_in[0];
    const int*   src      = (const int*)d_in[1];
    const int*   dst      = (const int*)d_in[2];
    const int*   vnodes   = (const int*)d_in[3];
    const float* gat_W    = (const float*)d_in[4];
    const float* gat_al   = (const float*)d_in[5];
    const float* gat_ar   = (const float*)d_in[6];
    const float* gat_b    = (const float*)d_in[7];
    const float* sem_W1   = (const float*)d_in[8];
    const float* sem_b1   = (const float*)d_in[9];
    const float* sem_w2   = (const float*)d_in[10];
    const float* time_W1  = (const float*)d_in[11];
    const float* time_b1  = (const float*)d_in[12];
    const float* time_w2  = (const float*)d_in[13];
    const float* pred_W   = (const float*)d_in[14];
    const float* pred_b   = (const float*)d_in[15];
    float* out = (float*)d_out;

    // ---- workspace carve-up (256B aligned) ----
    char* base = (char*)d_ws;
    size_t off = 0;
    auto alloc = [&](size_t bytes) -> char* {
        char* p = base + off;
        off = (off + bytes + 255) & ~(size_t)255;
        return p;
    };
    unsigned short* Wbf      = (unsigned short*)alloc((size_t)M_ * IN_ * EMB_ * 2);
    unsigned short* semW1bf  = (unsigned short*)alloc((size_t)EMB_ * HID_ * 2);
    unsigned short* timeW1bf = (unsigned short*)alloc((size_t)EMB_ * HID_ * 2);
    unsigned short* predWbf  = (unsigned short*)alloc((size_t)EMB_ * OUT_ * 2);
    float* proj = (float*)alloc((size_t)NV_ * EMB_ * 4);
    float* gout = (float*)alloc((size_t)NV_ * EMB_ * 4);
    float* el   = (float*)alloc((size_t)NV_ * H_ * 4);
    float* er   = (float*)alloc((size_t)NV_ * H_ * 4);
    float* mx   = (float*)alloc((size_t)NV_ * H_ * 4);
    float* den  = (float*)alloc((size_t)NV_ * H_ * 4);
    unsigned short* zbf    = (unsigned short*)alloc((size_t)NV_ * M_ * EMB_ * 2);
    unsigned short* ztbf   = (unsigned short*)alloc((size_t)NN_ * T_ * EMB_ * 2);
    unsigned short* tempbf = (unsigned short*)alloc((size_t)NN_ * EMB_ * 2);
    float* wlog  = (float*)alloc(256);       // [0..3]=logits, [4..7]=beta
    float* beta  = wlog + 4;

    const int FB = 4096;                     // blocks for grid-stride fills/elementwise

    // ---- one-time per call: bf16 weight copies; zero zt ----
    cast_bf16<<<FB, 256, 0, stream>>>(gat_W,   Wbf,      (size_t)M_ * IN_ * EMB_);
    cast_bf16<<<FB, 256, 0, stream>>>(sem_W1,  semW1bf,  (size_t)EMB_ * HID_);
    cast_bf16<<<FB, 256, 0, stream>>>(time_W1, timeW1bf, (size_t)EMB_ * HID_);
    cast_bf16<<<FB, 256, 0, stream>>>(pred_W,  predWbf,  (size_t)EMB_ * OUT_);
    fill_u32<<<FB, 256, 0, stream>>>((unsigned int*)ztbf, (size_t)NN_ * T_ * EMB_ / 2, 0u);

    const long long EH = (long long)E_ * H_;
    const size_t NVE = (size_t)NV_ * EMB_;

    for (int t = 0; t < T_; ++t) {
        for (int m = 0; m < M_; ++m) {
            const int*   s_tm = src + ((size_t)t * M_ + m) * E_;
            const int*   d_tm = dst + ((size_t)t * M_ + m) * E_;
            const float* al_m = gat_al + (size_t)m * EMB_;
            const float* ar_m = gat_ar + (size_t)m * EMB_;
            const float* b_m  = gat_b  + (size_t)m * EMB_;

            fill_u32<<<FB, 256, 0, stream>>>((unsigned int*)gout, NVE, 0u);
            fill_u32<<<64, 256, 0, stream>>>((unsigned int*)mx,  (size_t)NV_ * H_, 0xFF800000u);
            fill_u32<<<64, 256, 0, stream>>>((unsigned int*)den, (size_t)NV_ * H_, 0u);

            // proj = feat_t @ W_m   (A f32 -> bf16 in LDS)
            dim3 g(EMB_ / 128, NV_ / 16);
            wmma_gemm<true, 0><<<g, 256, 0, stream>>>(
                features + (size_t)t * NV_ * IN_, Wbf + (size_t)m * IN_ * EMB_,
                proj, IN_, EMB_, nullptr, nullptr, nullptr, 1);

            elr_kernel<<<1280, 256, 0, stream>>>(proj, al_m, ar_m, el, er, NV_ * H_);
            edge_max<<<8192, 256, 0, stream>>>(s_tm, d_tm, el, er, mx, EH);
            edge_den<<<8192, 256, 0, stream>>>(s_tm, d_tm, el, er, mx, den, EH);
            edge_scatter<<<20000, 256, 0, stream>>>(s_tm, d_tm, el, er, mx, den, proj, gout, E_);
            gat_finalize<<<FB, 256, 0, stream>>>(gout, b_m, zbf, m, NVE);
        }
        // semantic attention over the M paths
        fill_u32<<<1, 64, 0, stream>>>((unsigned int*)wlog, 4, 0u);
        dim3 gs(1, NV_ * M_ / 16);
        wmma_gemm<false, 1><<<gs, 256, 0, stream>>>(
            zbf, semW1bf, nullptr, EMB_, HID_, sem_b1, sem_w2, wlog, M_);
        beta_kernel<<<1, 32, 0, stream>>>(wlog, beta, M_, 1.0f / (float)NV_);
        emb_scatter<<<FB, 256, 0, stream>>>(zbf, beta, vnodes + (size_t)t * NV_, ztbf, t, NVE);
    }

    // temporal attention over T snapshots (all NN nodes, zero rows included)
    fill_u32<<<1, 64, 0, stream>>>((unsigned int*)wlog, 4, 0u);
    dim3 gt(1, NN_ * T_ / 16);
    wmma_gemm<false, 1><<<gt, 256, 0, stream>>>(
        ztbf, timeW1bf, nullptr, EMB_, HID_, time_b1, time_w2, wlog, T_);
    beta_kernel<<<1, 32, 0, stream>>>(wlog, beta, T_, 1.0f / (float)NN_);
    temporal_mix<<<FB, 256, 0, stream>>>(ztbf, beta, tempbf, (size_t)NN_ * EMB_);

    // final prediction: out = temporal @ pred_W + pred_b
    dim3 gp(1, NN_ / 16);
    wmma_gemm<false, 0><<<gp, 256, 0, stream>>>(
        tempbf, predWbf, out, EMB_, OUT_, pred_b, nullptr, nullptr, 1);
}
